// GRU_CRF_47545287967585
// MI455X (gfx1250) — matrix operand order repaired and checked by reading, not compiled, not run on hardware
//
#include <hip/hip_runtime.h>
#include <hip/hip_bf16.h>
#include <math.h>

// Problem sizes fixed by the reference
#define BATCH 128
#define TLEN  1024
#define DIN   128
#define HID   64
#define G3    192   // 3*HID
#define KTAG  32
#define MTILES_PER_WAVE 8   // xproj: M tiles processed per wave under one B-frag set

typedef __attribute__((ext_vector_type(16))) __bf16 v16bf;
typedef __attribute__((ext_vector_type(8)))  float  v8f;

__device__ __forceinline__ v8f wmma_bf16(v16bf a, v16bf b, v8f c) {
  // D = A(16x32 bf16) x B(32x16 bf16) + C(16x16 f32)
  return __builtin_amdgcn_wmma_f32_16x16x32_bf16(false, a, false, b, (short)0, c,
                                                 false, false);
}

// A-fragment (16x32 bf16) per ISA 7.12.2: lanes 0-15 hold row M=lane with
// K = k0 + {0..7,16..23}; lanes 16-31 hold row M=lane-16 with K = k0 + {8..15,24..31}.
__device__ __forceinline__ v16bf load_afrag(const float* base, int m, int hlf,
                                            int k0, int ld) {
  v16bf a;
#pragma unroll
  for (int e = 0; e < 16; ++e) {
    int k = k0 + ((e & 7) | ((e & 8) << 1)) + hlf * 8;
    a[e] = (__bf16)base[m * ld + k];
  }
  return a;
}

// B-fragment (32x16 bf16): lanes 0-15 hold col N=n0+lane, K=k0+{0..7,16..23};
// lanes 16-31 hold col N=n0+lane-16, K=k0+{8..15,24..31}.
__device__ __forceinline__ v16bf load_bfrag(const float* W, int k0, int n0,
                                            int m, int hlf, int ld) {
  v16bf b;
#pragma unroll
  for (int e = 0; e < 16; ++e) {
    int k = k0 + ((e & 7) | ((e & 8) << 1)) + hlf * 8;
    b[e] = (__bf16)W[k * ld + n0 + m];
  }
  return b;
}

// ---------------------------------------------------------------------------
// Kernel 1: xproj = X @ gru_kernel + gru_bias[0], stored bf16 (halves scratch BW).
// Each wave owns one 16-wide N tile and sweeps 8 consecutive M tiles, keeping
// all four loop-invariant B-fragments (K=128) resident in VGPRs.
// ---------------------------------------------------------------------------
__global__ __launch_bounds__(256) void xproj_kernel(
    const float* __restrict__ X, const float* __restrict__ Wg,
    const float* __restrict__ gbias, __bf16* __restrict__ xproj) {
  const int NT = G3 / 16;  // 12 column tiles
  int wave = (blockIdx.x * blockDim.x + threadIdx.x) >> 5;
  int lane = threadIdx.x & 31;
  int hlf = lane >> 4, m = lane & 15;
  int mg = wave / NT;                       // M group: 8 tiles = 128 rows
  int n0 = (wave % NT) * 16;
  int mbase = mg * (16 * MTILES_PER_WAVE);

  // Loop-invariant weight fragments for the whole K=128 reduction
  v16bf b0 = load_bfrag(Wg, 0,  n0, m, hlf, G3);
  v16bf b1 = load_bfrag(Wg, 32, n0, m, hlf, G3);
  v16bf b2 = load_bfrag(Wg, 64, n0, m, hlf, G3);
  v16bf b3 = load_bfrag(Wg, 96, n0, m, hlf, G3);
  float bias = gbias[n0 + m];

  for (int mt = 0; mt < MTILES_PER_WAVE; ++mt) {
    int m0 = mbase + mt * 16;
    // Prefetch the next tile's row for this lane (global_prefetch_b8)
    if (mt + 1 < MTILES_PER_WAVE)
      __builtin_prefetch(&X[(long)(m0 + 16 + m) * DIN], 0, 1);
    v8f c = {0.f, 0.f, 0.f, 0.f, 0.f, 0.f, 0.f, 0.f};
    v16bf a0 = load_afrag(X, m0 + m, hlf, 0, DIN);
    v16bf a1 = load_afrag(X, m0 + m, hlf, 32, DIN);
    v16bf a2 = load_afrag(X, m0 + m, hlf, 64, DIN);
    v16bf a3 = load_afrag(X, m0 + m, hlf, 96, DIN);
    c = wmma_bf16(a0, b0, c);
    c = wmma_bf16(a1, b1, c);
    c = wmma_bf16(a2, b2, c);
    c = wmma_bf16(a3, b3, c);
#pragma unroll
    for (int r = 0; r < 8; ++r) {
      int row = m0 + r + hlf * 8;  // C layout: VGPR r -> M=r (lanes<16), M=8+r
      xproj[(long)row * G3 + n0 + m] = (__bf16)(c[r] + bias);
    }
  }
}

// ---------------------------------------------------------------------------
// Kernel 2: per-batch sequence length (mask row sum) -> output + int scratch
// ---------------------------------------------------------------------------
__global__ void seqlen_kernel(const int* __restrict__ mask,
                              float* __restrict__ out_sl, int* __restrict__ sl_i) {
  int b = threadIdx.x;
  if (b < BATCH) {
    int s = 0;
    const int* row = mask + b * TLEN;
    for (int t = 0; t < TLEN; t += 4) s += row[t] + row[t + 1] + row[t + 2] + row[t + 3];
    out_sl[b] = (float)s;
    sl_i[b] = s;
  }
}

// Kernel 3: chain_kernel is returned verbatim by the reference
__global__ void chaincopy_kernel(const float* __restrict__ chain,
                                 float* __restrict__ outc) {
  outc[threadIdx.x] = chain[threadIdx.x];
}

// ---------------------------------------------------------------------------
// Kernel 4: fused GRU recurrence + emission GEMM + CRF Viterbi forward.
// 8 workgroups x 384 threads (12 waves); each owns 16 batch rows for all T.
// Recurrent/dense bf16 B-fragments are loop-invariant and live in VGPRs.
// ---------------------------------------------------------------------------
__global__ __launch_bounds__(384) void gru_crf_kernel(
    const int* __restrict__ mask, const __bf16* __restrict__ xproj,
    const float* __restrict__ Wr, const float* __restrict__ gbias,
    const float* __restrict__ Wd, const float* __restrict__ dbias,
    const float* __restrict__ chain, const float* __restrict__ lbound,
    const float* __restrict__ rbound, const int* __restrict__ sl_i,
    float* __restrict__ out_pot, unsigned char* __restrict__ bp,
    float* __restrict__ fscore) {
  __shared__ float sh_h[2][16 * HID];       // fp32 master hidden state (ping-pong)
  __shared__ float sh_rec[16 * G3];         // recurrent projection + rec bias
  __shared__ float sh_pot[16 * KTAG];       // emissions for this step
  __shared__ float sh_score[2][16 * KTAG];  // Viterbi scores (ping-pong)
  __shared__ float sh_chain[KTAG * KTAG];
  __shared__ int sh_sl[16];

  const int tid = threadIdx.x;
  const int lane = tid & 31;
  const int w = tid >> 5;  // wave 0..11
  const int hlf = lane >> 4;
  const int m = lane & 15;
  const int b0 = blockIdx.x * 16;

  for (int i = tid; i < 16 * HID; i += blockDim.x) sh_h[0][i] = 0.f;
  for (int i = tid; i < KTAG * KTAG; i += blockDim.x) sh_chain[i] = chain[i];
  if (tid < 16) sh_sl[tid] = sl_i[b0 + tid];

  // Loop-invariant B fragments: recurrent (each wave owns a 16-col tile of 192)
  const int n0r = w * 16;
  v16bf br0 = load_bfrag(Wr, 0, n0r, m, hlf, G3);
  v16bf br1 = load_bfrag(Wr, 32, n0r, m, hlf, G3);
  // Dense emission fragments (used by waves 0/1)
  const int n0d = (w & 1) * 16;
  v16bf bd0 = load_bfrag(Wd, 0, n0d, m, hlf, KTAG);
  v16bf bd1 = load_bfrag(Wd, 32, n0d, m, hlf, KTAG);

  const float* gb1 = gbias + G3;  // recurrent bias row
  __syncthreads();

  for (int t = 0; t < TLEN; ++t) {
    const float* hc = sh_h[t & 1];
    float* hnew = sh_h[(t + 1) & 1];

    // --- recurrent GEMM: rec = h @ Wr (12 waves x [16,64]x[64,16]) ---
    {
      v16bf a0 = load_afrag(hc, m, hlf, 0, HID);
      v16bf a1 = load_afrag(hc, m, hlf, 32, HID);
      v8f c = {0.f, 0.f, 0.f, 0.f, 0.f, 0.f, 0.f, 0.f};
      c = wmma_bf16(a0, br0, c);
      c = wmma_bf16(a1, br1, c);
#pragma unroll
      for (int r = 0; r < 8; ++r) {
        int lb = r + hlf * 8;
        int col = n0r + m;
        sh_rec[lb * G3 + col] = c[r] + gb1[col];
      }
    }
    __syncthreads();

    // --- fp32 gate math (Keras reset_after=True ordering z/r/h) ---
    for (int i = tid; i < 16 * HID; i += blockDim.x) {
      int b = i >> 6, j = i & (HID - 1);
      long xrow = (long)((b0 + b) * TLEN + t) * G3;
      float xz = (float)xproj[xrow + j];
      float xr = (float)xproj[xrow + HID + j];
      float xh = (float)xproj[xrow + 2 * HID + j];
      // Prefetch next timestep's projection row while we do transcendentals
      if (t + 1 < TLEN) __builtin_prefetch(&xproj[xrow + G3 + j], 0, 1);
      float rz = sh_rec[b * G3 + j];
      float rr = sh_rec[b * G3 + HID + j];
      float rh = sh_rec[b * G3 + 2 * HID + j];
      float z = 1.f / (1.f + __expf(-(xz + rz)));
      float rg = 1.f / (1.f + __expf(-(xr + rr)));
      float hh = tanhf(xh + rg * rh);
      float hp = hc[i];
      float hv = z * hp + (1.f - z) * hh;
      hnew[i] = mask[(b0 + b) * TLEN + t] ? hv : hp;  // masked steps carry state
    }
    __syncthreads();

    // --- emission GEMM + boundary energies (waves 0,1: [16,64]x[64,32]) ---
    if (w < 2) {
      v16bf a0 = load_afrag(hnew, m, hlf, 0, HID);
      v16bf a1 = load_afrag(hnew, m, hlf, 32, HID);
      v8f c = {0.f, 0.f, 0.f, 0.f, 0.f, 0.f, 0.f, 0.f};
      c = wmma_bf16(a0, bd0, c);
      c = wmma_bf16(a1, bd1, c);
#pragma unroll
      for (int r = 0; r < 8; ++r) {
        int lb = r + hlf * 8;
        int col = n0d + m;
        float p = c[r] + dbias[col];
        if (t == 0) p += lbound[col];
        if (t == sh_sl[lb] - 1) p += rbound[col];
        sh_pot[lb * KTAG + col] = p;
        out_pot[(long)((b0 + lb) * TLEN + t) * KTAG + col] = p;
      }
    }
    __syncthreads();

    // --- Viterbi forward step (max-plus DP with byte backpointers) ---
    {
      const float* sc = sh_score[t & 1];
      float* sn = sh_score[(t + 1) & 1];
      if (t == 0) {
        for (int i = tid; i < 16 * KTAG; i += blockDim.x) sn[i] = sh_pot[i];
      } else {
        for (int i = tid; i < 16 * KTAG; i += blockDim.x) {
          int b = i >> 5, kto = i & (KTAG - 1);
          float best = -3.4e38f;
          int arg = 0;
#pragma unroll
          for (int kf = 0; kf < KTAG; ++kf) {
            float v = sc[b * KTAG + kf] + sh_chain[kf * KTAG + kto];
            if (v > best) { best = v; arg = kf; }  // first-max, like jnp.argmax
          }
          sn[i] = best + sh_pot[i];
          bp[(long)(t - 1) * BATCH * KTAG + (b0 + b) * KTAG + kto] = (unsigned char)arg;
        }
      }
    }
    __syncthreads();
  }

  for (int i = tid; i < 16 * KTAG; i += blockDim.x)
    fscore[(b0 + (i >> 5)) * KTAG + (i & (KTAG - 1))] = sh_score[TLEN & 1][i];
}

// ---------------------------------------------------------------------------
// Kernel 5: per-batch backtrace through the byte backpointers
// ---------------------------------------------------------------------------
__global__ void backtrace_kernel(const float* __restrict__ fscore,
                                 const unsigned char* __restrict__ bp,
                                 float* __restrict__ decoded) {
  int b = threadIdx.x;
  if (b >= BATCH) return;
  const float* fs = fscore + b * KTAG;
  int tag = 0;
  float best = fs[0];
  for (int k = 1; k < KTAG; ++k)
    if (fs[k] > best) { best = fs[k]; tag = k; }
  decoded[b * TLEN + (TLEN - 1)] = (float)tag;
  for (int t = TLEN - 2; t >= 0; --t) {
    tag = bp[(long)t * BATCH * KTAG + b * KTAG + tag];
    decoded[b * TLEN + t] = (float)tag;
  }
}

extern "C" void kernel_launch(void* const* d_in, const int* in_sizes, int n_in,
                              void* d_out, int out_size, void* d_ws, size_t ws_size,
                              hipStream_t stream) {
  (void)in_sizes; (void)n_in; (void)out_size; (void)ws_size;
  const float* X      = (const float*)d_in[0];
  const int*   mask   = (const int*)d_in[1];
  const float* Wg     = (const float*)d_in[2];  // [128,192]
  const float* Wr     = (const float*)d_in[3];  // [64,192]
  const float* gbias  = (const float*)d_in[4];  // [2,192]
  const float* Wd     = (const float*)d_in[5];  // [64,32]
  const float* dbias  = (const float*)d_in[6];  // [32]
  const float* chain  = (const float*)d_in[7];  // [32,32]
  const float* lbound = (const float*)d_in[8];
  const float* rbound = (const float*)d_in[9];

  // Output tuple, flat in return order: decoded[B,T], potentials[B,T,K],
  // seq_len[B], chain_kernel[K,K]
  float* out = (float*)d_out;
  float* out_dec = out;
  float* out_pot = out + (long)BATCH * TLEN;
  float* out_sl  = out_pot + (long)BATCH * TLEN * KTAG;
  float* out_ch  = out_sl + BATCH;

  // Scratch layout (~55 MB total)
  char* wsp = (char*)d_ws;
  size_t off = 0;
  __bf16* xproj = (__bf16*)(wsp + off);
  off += (size_t)BATCH * TLEN * G3 * sizeof(__bf16);          // 50.3 MB
  unsigned char* bp = (unsigned char*)(wsp + off);
  off += (size_t)TLEN * BATCH * KTAG;                         // 4.2 MB
  float* fscore = (float*)(wsp + off);
  off += (size_t)BATCH * KTAG * sizeof(float);
  int* sl_i = (int*)(wsp + off);

  seqlen_kernel<<<1, BATCH, 0, stream>>>(mask, out_sl, sl_i);
  chaincopy_kernel<<<1, KTAG * KTAG, 0, stream>>>(chain, out_ch);
  // (B*T / (16*MTILES_PER_WAVE)) M-groups x 12 N-tiles waves, 8 waves/block
  xproj_kernel<<<(BATCH * TLEN / (16 * MTILES_PER_WAVE)) * (G3 / 16) / 8, 256, 0,
                stream>>>(X, Wg, gbias, xproj);
  gru_crf_kernel<<<BATCH / 16, 384, 0, stream>>>(mask, xproj, Wr, gbias, Wd, dbias,
                                                 chain, lbound, rbound, sl_i, out_pot,
                                                 bp, fscore);
  backtrace_kernel<<<1, BATCH, 0, stream>>>(fscore, bp, out_dec);
}
